// Patcher_21620865368362
// MI455X (gfx1250) — compile-verified, matrix operand order checked by reading, not run on hardware
//
#include <hip/hip_runtime.h>

#define BOX   64
#define PROC  128
#define NIMG  8
#define NBOX  8
#define IMG_H 1024
#define IMG_W 1024
#define CH    3
#define IMG_ELEMS (IMG_H * IMG_W * CH)      // 3,145,728 per image
#define PADW      (PROC + 2)                // halo-padded pb width
#define PAD_ELEMS (PADW * PADW * CH)        // 50,700 halves
#define ACC_ELEMS (BOX * BOX * CH)          // 12,288
#define NCHUNK    ((PROC * PROC) / 16)      // 1024 WMMA chunks per box

typedef __attribute__((ext_vector_type(16))) _Float16 v16h;
typedef __attribute__((ext_vector_type(8)))  float    v8f;

// ---------------------------------------------------------------------------
// Kernel 1: stream images -> out (bandwidth-bound part, ~200MB @ 23.3TB/s)
// ---------------------------------------------------------------------------
__global__ __launch_bounds__(256) void copy_images(const float4* __restrict__ src,
                                                   float4* __restrict__ dst, int n4) {
    int i = blockIdx.x * blockDim.x + threadIdx.x;
    if (i + 8192 < n4) __builtin_prefetch(&src[i + 8192], 0, 1);
    if (i < n4) dst[i] = src[i];
}

// Branch-free tanh: sign(x) * (1-e)/(1+e), e = exp(-2|x|).
__device__ __forceinline__ float fast_tanh(float x) {
    float ax = fabsf(x);
    float e  = __expf(-2.f * ax);                       // v_exp_f32
    float y  = (1.f - e) * __builtin_amdgcn_rcpf(1.f + e);
    return copysignf(y, x);
}

// ---------------------------------------------------------------------------
// Kernel 2: one block per image; sequential scan over its 8 boxes.
//   upsample (global -> halo-padded LDS f16) -> WMMA conv + tanh + loss +
//   branch-free antialiased downsample scatter -> clipped global write-back.
// LDS: 101,400B pb(f16, 130x130x3) + 49,152B accumulator(f32) ~= 147KB / WG.
// ---------------------------------------------------------------------------
__global__ __launch_bounds__(256) void patch_kernel(float* __restrict__ out,
                                                    const float* __restrict__ W,    // (3,3,3,3) HWIO
                                                    const int* __restrict__ box_yx, // (NIMG,NBOX,2)
                                                    float* __restrict__ losses) {   // NIMG f32 (ws)
    __shared__ _Float16 s_pb[PAD_ELEMS];
    __shared__ float    s_acc[ACC_ELEMS];
    __shared__ float    s_loss;

    const int img    = blockIdx.x;
    const int tid    = threadIdx.x;
    const int lane   = tid & 31;
    const int wave   = tid >> 5;
    const int laneHi = lane >> 4;
    float* image = out + (size_t)img * IMG_ELEMS;

    if (tid == 0) s_loss = 0.f;

    // B fragment (weights, K=32 x N=16; rows k>=27 and cols n>=3 are ZERO) and
    // per-lane A-tap offsets (fixed per (j,laneHi)); built once, registers.
    // Because B's dead K-rows are zero, A needs NO masking: the clamped tap
    // (kk=min(k,26)) loads valid finite data whose product contribution is 0.
    v16h b_frag;
    int  aoff[16];
    {
        const int col = lane & 15;
#pragma unroll
        for (int j = 0; j < 16; ++j) {
            int k  = j + ((j >> 3) << 3) + (laneHi << 3);   // ISA 16-bit striping
            int kk = min(k, 26);
            int ky = kk / 9, kx = (kk / 3) % 3, ci = kk % 3;
            aoff[j] = (ky * PADW + kx) * CH + ci;           // halo: always in-bounds
            float wv = 0.f;
            if (k < 27 && col < CH) wv = W[k * CH + col];
            b_frag[j] = (_Float16)wv;
        }
    }

    float acc = 0.f;  // raw squared-error accumulator across all boxes

    for (int b = 0; b < NBOX; ++b) {
        const int by = box_yx[(img * NBOX + b) * 2 + 0];
        const int bx = box_yx[(img * NBOX + b) * 2 + 1];

        // ---- upsample crop 64->128 (half-pixel bilinear) into padded LDS;
        //      halo ring stays zero (SAME conv zero-padding). Zero s_acc too.
        for (int idx = tid; idx < PAD_ELEMS; idx += 256) {
            int c  = idx % CH;
            int p  = idx / CH;
            int qx = p % PADW;
            int qy = p / PADW;
            float v = 0.f;
            if (qx >= 1 && qx <= PROC && qy >= 1 && qy <= PROC) {
                int px = qx - 1, py = qy - 1;
                float sy = 0.5f * py - 0.25f;
                float sx = 0.5f * px - 0.25f;
                int y0 = (int)floorf(sy), x0 = (int)floorf(sx);
                float wy = sy - y0, wx = sx - x0;
                int y0c = max(0, min(BOX - 1, y0));
                int y1c = max(0, min(BOX - 1, y0 + 1));
                int x0c = max(0, min(BOX - 1, x0));
                int x1c = max(0, min(BOX - 1, x0 + 1));
                const float* r0 = &image[((by + y0c) * IMG_W + bx) * CH + c];
                const float* r1 = &image[((by + y1c) * IMG_W + bx) * CH + c];
                float v00 = r0[x0c * CH], v01 = r0[x1c * CH];
                float v10 = r1[x0c * CH], v11 = r1[x1c * CH];
                v = (1.f - wy) * ((1.f - wx) * v00 + wx * v01)
                  +        wy  * ((1.f - wx) * v10 + wx * v11);
            }
            s_pb[idx] = (_Float16)v;
        }
        for (int idx = tid; idx < ACC_ELEMS; idx += 256) s_acc[idx] = 0.f;
        __syncthreads();

        // ---- 3x3 SAME conv (3->3) as im2col GEMM on v_wmma_f32_16x16x32_f16
        for (int chunk = wave; chunk < NCHUNK; chunk += 8) {
            const int m    = chunk * 16 + (lane & 15);   // this lane's pixel row
            const int py   = m >> 7;
            const int px   = m & 127;
            const int base = (py * PADW + px) * CH;      // tap (ky,kx)=(0,0) in padded pb
            v16h a;
#pragma unroll
            for (int j = 0; j < 16; ++j)
                a[j] = s_pb[base + aoff[j]];             // unconditional, unmasked
            v8f c = {};
            c = __builtin_amdgcn_wmma_f32_16x16x32_f16(
                    false, a, false, b_frag, (short)0, c, false, false);

            const int ch = lane & 15;  // D: col = output channel
            if (ch < CH) {
#pragma unroll
                for (int r = 0; r < 8; ++r) {
                    int pix = chunk * 16 + r + 8 * laneHi;   // D row -> pixel
                    int pyo = pix >> 7, pxo = pix & 127;
                    float v   = fast_tanh(c[r]);
                    float pbv = (float)s_pb[((pyo + 1) * PADW + (pxo + 1)) * CH + ch];
                    float d   = v - pbv;
                    acc += d * d;
                    // antialiased 2x downsample scatter: each source pixel hits
                    // exactly 2 outputs per dim; OOB secondary folds onto main
                    // (reproduces edge-clamped weights). Constant weights.
                    int oym = pyo >> 1;
                    int oys = oym + ((pyo & 1) ? 1 : -1);
                    oys = ((unsigned)oys < BOX) ? oys : oym;
                    int oxm = pxo >> 1;
                    int oxs = oxm + ((pxo & 1) ? 1 : -1);
                    oxs = ((unsigned)oxs < BOX) ? oxs : oxm;
                    atomicAdd(&s_acc[(oym * BOX + oxm) * CH + ch], 0.140625f * v);
                    atomicAdd(&s_acc[(oym * BOX + oxs) * CH + ch], 0.046875f * v);
                    atomicAdd(&s_acc[(oys * BOX + oxm) * CH + ch], 0.046875f * v);
                    atomicAdd(&s_acc[(oys * BOX + oxs) * CH + ch], 0.015625f * v);
                }
            }
        }
        __syncthreads();

        // ---- clipped write-back of the 64x64x3 patch into the image
        for (int idx = tid; idx < ACC_ELEMS; idx += 256) {
            int c  = idx % CH;
            int p  = idx / CH;
            int ox = p % BOX;
            int oy = p / BOX;
            float v = s_acc[idx];
            v = fminf(1.f, fmaxf(-1.f, v));
            image[((by + oy) * IMG_W + (bx + ox)) * CH + c] = v;
        }
        __threadfence();   // make write-back visible to next box's crop reads
        __syncthreads();
    }

    atomicAdd(&s_loss, acc);
    __syncthreads();
    if (tid == 0) losses[img] = s_loss / (float)(PROC * PROC * CH);
}

// ---------------------------------------------------------------------------
// Kernel 3: 8-element cumulative sum of per-image losses -> tail of d_out
// ---------------------------------------------------------------------------
__global__ void cumsum_losses(const float* __restrict__ losses,
                              float* __restrict__ out_tail) {
    if (blockIdx.x == 0 && threadIdx.x == 0) {
        float s = 0.f;
        for (int i = 0; i < NIMG; ++i) { s += losses[i]; out_tail[i] = s; }
    }
}

extern "C" void kernel_launch(void* const* d_in, const int* in_sizes, int n_in,
                              void* d_out, int out_size, void* d_ws, size_t ws_size,
                              hipStream_t stream) {
    const float* images = (const float*)d_in[0];  // (8,1024,1024,3) f32
    const float* W      = (const float*)d_in[1];  // (3,3,3,3) f32
    const int*   box_yx = (const int*)  d_in[2];  // (8,8,2) i32
    float* out    = (float*)d_out;
    float* losses = (float*)d_ws;                  // 8 floats of scratch

    const int n4 = (NIMG * IMG_ELEMS) / 4;         // 6,291,456 float4
    copy_images<<<(n4 + 255) / 256, 256, 0, stream>>>(
        (const float4*)images, (float4*)out, n4);

    patch_kernel<<<NIMG, 256, 0, stream>>>(out, W, box_yx, losses);

    cumsum_losses<<<1, 32, 0, stream>>>(losses, out + (size_t)NIMG * IMG_ELEMS);
}
